// RPN_68719476820
// MI455X (gfx1250) — matrix operand order repaired and checked by reading, not compiled, not run on hardware
//
#include <hip/hip_runtime.h>

// ---------------- types for WMMA ----------------
typedef __attribute__((ext_vector_type(16))) __bf16 v16bf;
typedef __attribute__((ext_vector_type(8)))  float  v8f;

#define CATN 4192          // 4*1000 + 192 (level4: 8*8*3=192)
#define NWORDS 131         // CATN/32
#define PADN 8192
#define SCALE_CLAMP 4.135166556742356f

__device__ __forceinline__ unsigned short f32_to_bf16(float f) {
  unsigned int u = __float_as_uint(f);
  unsigned int r = u + 0x7FFFu + ((u >> 16) & 1u); // round-to-nearest-even
  return (unsigned short)(r >> 16);
}
__device__ __forceinline__ unsigned int pack_bf16x2(float lo, float hi) {
  return (unsigned int)f32_to_bf16(lo) | ((unsigned int)f32_to_bf16(hi) << 16);
}

__device__ __forceinline__ unsigned int ord_f32(float f) {
  unsigned int u = __float_as_uint(f);
  return (u & 0x80000000u) ? ~u : (u | 0x80000000u);
}
__device__ __forceinline__ float unord_f32(unsigned int u) {
  return __uint_as_float((u & 0x80000000u) ? (u ^ 0x80000000u) : ~u);
}

// ---------------- zero fill ----------------
__global__ void k_zero(float* p, int n) {
  int i = blockIdx.x * blockDim.x + threadIdx.x;
  if (i < n) p[i] = 0.f;
}

// ---------------- conv weights fp32 OIHW -> bf16 [o][tap*256+cin] ----------------
__global__ void k_cvt_w(const float* __restrict__ conv_w, unsigned short* __restrict__ wbf) {
  int idx = blockIdx.x * blockDim.x + threadIdx.x;
  if (idx >= 256 * 2304) return;
  int o = idx / 2304, rem = idx - o * 2304;
  int tap = rem >> 8, cin = rem & 255;
  int ky = tap / 3, kx = tap - ky * 3;
  wbf[idx] = f32_to_bf16(conv_w[(((size_t)o * 256 + cin) * 3 + ky) * 3 + kx]);
}

// =====================================================================
// Implicit-GEMM 3x3 conv via WMMA bf16.
//   A: im2col patches [M=2*H*W][K=2304], k = tap*256+cin
//   B: wbf [N=256][K]
// Workgroup: 256 threads (8 waves). WG tile 64(M) x 128(N).
// Wave tile: 32x32 -> 4 accumulators, 4 WMMAs per K step.
// LDS double-buffered; fragment-ready swizzled layout so each lane's
// fragment is 8 consecutive dwords (-> ds_load_b128).
// =====================================================================
template<int H, int W>
__global__ __launch_bounds__(256)
void k_conv_wmma(const float* __restrict__ feat,         // [2][256][H][W]
                 const unsigned short* __restrict__ wbf, // [256][2304]
                 const float* __restrict__ conv_b,       // [256]
                 float* __restrict__ y)                   // [M][256]
{
  constexpr int HW = H * W;
  // fragment-ready LDS: sA[buf][(rb*32 + lane)*8 + j], rb = 16-row block (4)
  //                     sB[buf][(cb*32 + lane)*8 + j], cb = 16-col block (8)
  __shared__ unsigned int sA[2][4 * 32 * 8];   // 2 x 4KB
  __shared__ unsigned int sB[2][8 * 32 * 8];   // 2 x 8KB

  const int t    = threadIdx.x;
  const int lane = t & 31;
  const int wid  = t >> 5;
  const int rb0  = (wid >> 2) * 2;      // wave row-block pair: 0 or 2
  const int cb0  = (wid & 3) * 2;       // wave col-block pair: 0,2,4,6
  const int m_base = blockIdx.x * 64;
  const int n_base = blockIdx.y * 128;

  // ---- A staging coords: 64 rows x 16 kpairs; thread -> (row, 4 kpairs)
  const int arow = t >> 2;              // 0..63
  const int aq   = t & 3;               // kpair group: kp = aq*4 + q
  const int m_g  = m_base + arow;
  const int bimg = m_g / HW;            // compile-time power-of-two -> shift
  const int pix  = m_g - bimg * HW;
  const int hh   = pix / W;
  const int ww   = pix - hh * W;
  const int arb  = arow >> 4, am16 = arow & 15;

  // ---- B staging coords: 128 cols x 16 kpairs; thread -> (col, 8 kpairs)
  const int brow = t >> 1;              // 0..127
  const int bhalf = t & 1;              // kp = bhalf*8 + kpl
  const int bcb  = brow >> 4, bn16 = brow & 15;

  v8f acc00 = {0.f,0.f,0.f,0.f,0.f,0.f,0.f,0.f};
  v8f acc01 = {0.f,0.f,0.f,0.f,0.f,0.f,0.f,0.f};
  v8f acc10 = {0.f,0.f,0.f,0.f,0.f,0.f,0.f,0.f};
  v8f acc11 = {0.f,0.f,0.f,0.f,0.f,0.f,0.f,0.f};

  auto stage = [&](int k0, int buf) {
    // ---- A: im2col gather fp32 -> packed bf16 pairs, swizzled store ----
    {
      const int tap = k0 >> 8;                          // constant within slice
      const int dy = tap / 3 - 1, dx = tap - (tap / 3) * 3 - 1;
      const int h2 = hh + dy, w2 = ww + dx;
      const bool inb = (h2 >= 0) && (h2 < H) && (w2 >= 0) && (w2 < W);
      const int cin0 = (k0 & 255) + aq * 8;             // 8 consecutive cin
      const long base = (((long)bimg * 256 + cin0) * H + h2) * W + w2;
      #pragma unroll
      for (int q = 0; q < 4; ++q) {
        const int kp = aq * 4 + q;
        float v0 = inb ? feat[base + (long)(2 * q) * HW] : 0.f;
        float v1 = inb ? feat[base + (long)(2 * q + 1) * HW] : 0.f;
        // A layout: kp -> half = kp>>3, j = kp&7
        const int half = kp >> 3;
        const int j    = kp & 7;
        sA[buf][((arb * 32) + half * 16 + am16) * 8 + j] = pack_bf16x2(v0, v1);
      }
    }
    // ---- B: bf16 weights, vectorized uint4 loads, swizzled store ----
    {
      const unsigned int* srcu =
          (const unsigned int*)(wbf + (size_t)(n_base + brow) * 2304 + k0) + bhalf * 8;
      uint4 w0 = ((const uint4*)srcu)[0];
      uint4 w1 = ((const uint4*)srcu)[1];
      unsigned int uu[8] = {w0.x, w0.y, w0.z, w0.w, w1.x, w1.y, w1.z, w1.w};
      // B layout: kp -> half = kp>>3, j = kp&7 ; bhalf==half for this thread
      #pragma unroll
      for (int kpl = 0; kpl < 8; ++kpl)
        sB[buf][((bcb * 32) + bhalf * 16 + bn16) * 8 + kpl] = uu[kpl];
    }
  };

  stage(0, 0);
  __syncthreads();

  constexpr int NSTEP = 2304 / 32;   // 72
  for (int step = 0; step < NSTEP; ++step) {
    const int cur = step & 1;
    if (step + 1 < NSTEP) stage((step + 1) * 32, cur ^ 1);

    union F { v16bf v; unsigned int u[8]; };
    F a0, a1, b0, b1;
    #pragma unroll
    for (int j = 0; j < 8; ++j) {
      a0.u[j] = sA[cur][((rb0    ) * 32 + lane) * 8 + j];
      a1.u[j] = sA[cur][((rb0 + 1) * 32 + lane) * 8 + j];
      b0.u[j] = sB[cur][((cb0    ) * 32 + lane) * 8 + j];
      b1.u[j] = sB[cur][((cb0 + 1) * 32 + lane) * 8 + j];
    }
    acc00 = __builtin_amdgcn_wmma_f32_16x16x32_bf16(false, a0.v, false, b0.v,
                                                    (short)0, acc00, false, false);
    acc01 = __builtin_amdgcn_wmma_f32_16x16x32_bf16(false, a0.v, false, b1.v,
                                                    (short)0, acc01, false, false);
    acc10 = __builtin_amdgcn_wmma_f32_16x16x32_bf16(false, a1.v, false, b0.v,
                                                    (short)0, acc10, false, false);
    acc11 = __builtin_amdgcn_wmma_f32_16x16x32_bf16(false, a1.v, false, b1.v,
                                                    (short)0, acc11, false, false);
    __syncthreads();
  }

  // ---- epilogue: bias + ReLU, scatter per 16x16 f32 C layout ----
  const int lhalf = lane >> 4;
  const int l15   = lane & 15;
  v8f* accs[4] = {&acc00, &acc01, &acc10, &acc11};
  #pragma unroll
  for (int ti = 0; ti < 2; ++ti) {
    const int mrow0 = m_base + (rb0 + ti) * 16 + (lhalf << 3);
    #pragma unroll
    for (int tj = 0; tj < 2; ++tj) {
      const int n_out = n_base + (cb0 + tj) * 16 + l15;
      const float bias = conv_b[n_out];
      v8f& a = *accs[ti * 2 + tj];
      #pragma unroll
      for (int r = 0; r < 8; ++r) {
        float v = a[r] + bias;
        v = v > 0.f ? v : 0.f;
        y[(size_t)(mrow0 + r) * 256 + n_out] = v;
      }
    }
  }
}

// ---------------- 1x1 projections: scores (3) + deltas (12) per pixel ----------------
__global__ void k_project(const float* __restrict__ y,        // [2*HW][256]
                          const float* __restrict__ score_w,  // [3][256]
                          const float* __restrict__ score_b,  // [3]
                          const float* __restrict__ delta_w,  // [12][256]
                          const float* __restrict__ delta_b,  // [12]
                          float* __restrict__ scores,         // [2][HW*3]
                          float* __restrict__ deltas,         // [2][HW*3][4]
                          int HW)
{
  int m = blockIdx.x * blockDim.x + threadIdx.x;
  if (m >= 2 * HW) return;
  int b = m / HW, pix = m - b * HW;
  const float* yr = y + (size_t)m * 256;
  float s[3] = {0, 0, 0};
  float d[12] = {0, 0, 0, 0, 0, 0, 0, 0, 0, 0, 0, 0};
  for (int c = 0; c < 256; ++c) {
    float v = yr[c];
    #pragma unroll
    for (int a = 0; a < 3; ++a) s[a] += v * score_w[a * 256 + c];
    #pragma unroll
    for (int q = 0; q < 12; ++q) d[q] += v * delta_w[q * 256 + c];
  }
  const int n = HW * 3;
  for (int a = 0; a < 3; ++a) {
    int i = pix * 3 + a;
    scores[(size_t)b * n + i] = s[a] + score_b[a];
    #pragma unroll
    for (int j = 0; j < 4; ++j)
      deltas[((size_t)b * n + i) * 4 + j] = d[a * 4 + j] + delta_b[a * 4 + j];
  }
}

// ---------------- sorted top-K + anchor decode + clip ----------------
__global__ void k_topk_decode(float* __restrict__ sc,          // destructive [2][n]
                              const float* __restrict__ dl,    // [2][n][4]
                              float* __restrict__ cat_boxes,   // [2][CATN][4]
                              float* __restrict__ cat_scores,  // [2][CATN]
                              int n, int K, int offc,
                              int W, float astride, float asize)
{
  const int b = blockIdx.x;
  float* s = sc + (size_t)b * n;
  const float* dlt = dl + (size_t)b * n * 4;
  __shared__ unsigned long long red[1024];
  const int T = blockDim.x;

  for (int r = 0; r < K; ++r) {
    unsigned long long best = 0ull;
    for (int i = threadIdx.x; i < n; i += T) {
      unsigned long long pk = ((unsigned long long)ord_f32(s[i]) << 32)
                            | (unsigned long long)(0xFFFFFFFFu - (unsigned)i);
      best = pk > best ? pk : best;
    }
    red[threadIdx.x] = best;
    __syncthreads();
    for (int st = T >> 1; st > 0; st >>= 1) {
      if (threadIdx.x < st && red[threadIdx.x + st] > red[threadIdx.x])
        red[threadIdx.x] = red[threadIdx.x + st];
      __syncthreads();
    }
    if (threadIdx.x == 0) {
      unsigned long long pk = red[0];
      int idx = (int)(0xFFFFFFFFu - (unsigned)(pk & 0xFFFFFFFFull));
      float val = unord_f32((unsigned)(pk >> 32));
      // anchor from index: idx = (h*W+w)*3 + a
      int a = idx % 3, p = idx / 3;
      int ah = p / W, aw_ = p - ah * W;
      const float SQ[3] = {0.70710678118654752f, 1.0f, 1.41421356237309505f};
      float sq = SQ[a];
      float bw = asize / sq, bh = asize * sq;
      float cx = (aw_ + 0.5f) * astride, cy = (ah + 0.5f) * astride;
      float ddx = dlt[(size_t)idx * 4 + 0], ddy = dlt[(size_t)idx * 4 + 1];
      float ddw = dlt[(size_t)idx * 4 + 2], ddh = dlt[(size_t)idx * 4 + 3];
      float pcx = ddx * bw + cx, pcy = ddy * bh + cy;
      float pw = expf(fminf(ddw, SCALE_CLAMP)) * bw;
      float ph = expf(fminf(ddh, SCALE_CLAMP)) * bh;
      float x1 = fminf(fmaxf(pcx - 0.5f * pw, 0.f), 512.f);
      float y1 = fminf(fmaxf(pcy - 0.5f * ph, 0.f), 512.f);
      float x2 = fminf(fmaxf(pcx + 0.5f * pw, 0.f), 512.f);
      float y2 = fminf(fmaxf(pcy + 0.5f * ph, 0.f), 512.f);
      size_t o = (size_t)b * CATN + offc + r;
      cat_boxes[o * 4 + 0] = x1; cat_boxes[o * 4 + 1] = y1;
      cat_boxes[o * 4 + 2] = x2; cat_boxes[o * 4 + 3] = y2;
      cat_scores[o] = val;
      s[idx] = -3.0e38f;  // suppress for next round
    }
    __syncthreads();
  }
}

// ---------------- per-image bitonic sort (descending, stable tiebreak) ----------------
__global__ void k_sort(const float* __restrict__ cat_scores,
                       const float* __restrict__ cat_boxes,
                       float* __restrict__ sorted_boxes)   // [2][CATN][4]
{
  const int b = blockIdx.x;
  __shared__ unsigned long long key[PADN];   // 64 KB
  for (int i = threadIdx.x; i < PADN; i += blockDim.x) {
    unsigned long long pk = 0ull;
    if (i < CATN)
      pk = ((unsigned long long)ord_f32(cat_scores[(size_t)b * CATN + i]) << 32)
         | (unsigned long long)(0xFFFFFFFFu - (unsigned)i);
    key[i] = pk;
  }
  __syncthreads();
  for (int k = 2; k <= PADN; k <<= 1) {
    for (int j = k >> 1; j > 0; j >>= 1) {
      for (int i = threadIdx.x; i < PADN; i += blockDim.x) {
        int ixj = i ^ j;
        if (ixj > i) {
          bool up = ((i & k) == 0);   // ascending subsequence
          unsigned long long a = key[i], c = key[ixj];
          if (up ? (a > c) : (a < c)) { key[i] = c; key[ixj] = a; }
        }
      }
      __syncthreads();
    }
  }
  // j-th largest lives at PADN-1-j
  for (int j = threadIdx.x; j < CATN; j += blockDim.x) {
    unsigned long long pk = key[PADN - 1 - j];
    int idx = (int)(0xFFFFFFFFu - (unsigned)(pk & 0xFFFFFFFFull));
    #pragma unroll
    for (int q = 0; q < 4; ++q)
      sorted_boxes[((size_t)b * CATN + j) * 4 + q] =
          cat_boxes[((size_t)b * CATN + idx) * 4 + q];
  }
}

// ---------------- NMS suppression bitmask (32x32 tiles) ----------------
__global__ void k_nms_mask(const float* __restrict__ sorted_boxes,
                           unsigned int* __restrict__ mask)  // [2][CATN][NWORDS]
{
  const int b = blockIdx.z;
  const int row = blockIdx.x * 32 + threadIdx.x;
  const int colb = blockIdx.y;
  if (row >= CATN) return;
  const float* bp = sorted_boxes + (size_t)b * CATN * 4;
  float x1 = bp[row * 4], y1 = bp[row * 4 + 1], x2 = bp[row * 4 + 2], y2 = bp[row * 4 + 3];
  float area = fmaxf(x2 - x1, 0.f) * fmaxf(y2 - y1, 0.f);
  unsigned int bits = 0;
  for (int c = 0; c < 32; ++c) {
    int col = colb * 32 + c;
    if (col >= CATN || col <= row) continue;
    float u1 = bp[col * 4], v1 = bp[col * 4 + 1], u2 = bp[col * 4 + 2], v2 = bp[col * 4 + 3];
    float areac = fmaxf(u2 - u1, 0.f) * fmaxf(v2 - v1, 0.f);
    float ix = fminf(x2, u2) - fmaxf(x1, u1);
    float iy = fminf(y2, v2) - fmaxf(y1, v1);
    float inter = fmaxf(ix, 0.f) * fmaxf(iy, 0.f);
    float uni = area + areac - inter;
    if (inter / fmaxf(uni, 1e-9f) > 0.7f) bits |= (1u << c);
  }
  mask[((size_t)b * CATN + row) * NWORDS + colb] = bits;
}

// ---------------- greedy sequential scan, emit up to 1000 kept boxes ----------------
__global__ void k_nms_scan(const float* __restrict__ sorted_boxes,
                           const unsigned int* __restrict__ mask,
                           float* __restrict__ out)  // [2][1000][4], pre-zeroed
{
  const int b = blockIdx.x;
  __shared__ unsigned int removed[NWORDS];
  __shared__ int sh_rank;
  __shared__ int count;
  for (int t = threadIdx.x; t < NWORDS; t += blockDim.x) removed[t] = 0;
  if (threadIdx.x == 0) count = 0;
  __syncthreads();
  for (int i = 0; i < CATN; ++i) {
    if (threadIdx.x == 0) {
      if ((removed[i >> 5] >> (i & 31)) & 1u) sh_rank = -1;
      else { sh_rank = count; count++; }
    }
    __syncthreads();
    if (sh_rank >= 0) {
      if (sh_rank < 1000 && threadIdx.x < 4)
        out[((size_t)b * 1000 + sh_rank) * 4 + threadIdx.x] =
            sorted_boxes[((size_t)b * CATN + i) * 4 + threadIdx.x];
      const unsigned int* mrow = mask + ((size_t)b * CATN + i) * NWORDS;
      for (int t = threadIdx.x; t < NWORDS; t += blockDim.x) removed[t] |= mrow[t];
    }
    __syncthreads();
  }
}

// ---------------- orchestration ----------------
extern "C" void kernel_launch(void* const* d_in, const int* in_sizes, int n_in,
                              void* d_out, int out_size, void* d_ws, size_t ws_size,
                              hipStream_t stream) {
  const float* feats[5];
  for (int i = 0; i < 5; ++i) feats[i] = (const float*)d_in[i];
  const float* conv_w  = (const float*)d_in[5];
  const float* conv_b  = (const float*)d_in[6];
  const float* score_w = (const float*)d_in[7];
  const float* score_b = (const float*)d_in[8];
  const float* delta_w = (const float*)d_in[9];
  const float* delta_b = (const float*)d_in[10];
  float* out = (float*)d_out;

  // workspace carve-out
  char* ws = (char*)d_ws;
  size_t off = 0;
  auto carve = [&](size_t bytes) -> void* {
    off = (off + 255) & ~(size_t)255;
    void* p = ws + off;
    off += bytes;
    return p;
  };
  unsigned short* wbf       = (unsigned short*)carve((size_t)256 * 2304 * 2);
  float* y                  = (float*)carve((size_t)2 * 16384 * 256 * 4);   // max level
  float* sc                 = (float*)carve((size_t)2 * 49152 * 4);
  float* dl                 = (float*)carve((size_t)2 * 49152 * 4 * 4);
  float* cat_scores         = (float*)carve((size_t)2 * CATN * 4);
  float* cat_boxes          = (float*)carve((size_t)2 * CATN * 16);
  float* sorted_boxes       = (float*)carve((size_t)2 * CATN * 16);
  unsigned int* mask        = (unsigned int*)carve((size_t)2 * CATN * NWORDS * 4);
  (void)ws_size; (void)in_sizes; (void)n_in; (void)out_size;

  k_zero<<<(2 * 1000 * 4 + 255) / 256, 256, 0, stream>>>(out, 2 * 1000 * 4);
  k_cvt_w<<<(256 * 2304 + 255) / 256, 256, 0, stream>>>(conv_w, wbf);

  const int   Hs[5] = {128, 64, 32, 16, 8};
  const float STR[5] = {4.f, 8.f, 16.f, 32.f, 64.f};
  const float SIZ[5] = {32.f, 64.f, 128.f, 256.f, 512.f};
  int offc = 0;
  for (int L = 0; L < 5; ++L) {
    const int H = Hs[L], W = Hs[L], HW = H * W, M = 2 * HW, n = HW * 3;
    const int K = n < 1000 ? n : 1000;
    dim3 gconv(M / 64, 2);
    switch (L) {
      case 0: k_conv_wmma<128, 128><<<gconv, 256, 0, stream>>>(feats[L], wbf, conv_b, y); break;
      case 1: k_conv_wmma< 64,  64><<<gconv, 256, 0, stream>>>(feats[L], wbf, conv_b, y); break;
      case 2: k_conv_wmma< 32,  32><<<gconv, 256, 0, stream>>>(feats[L], wbf, conv_b, y); break;
      case 3: k_conv_wmma< 16,  16><<<gconv, 256, 0, stream>>>(feats[L], wbf, conv_b, y); break;
      default:k_conv_wmma<  8,   8><<<gconv, 256, 0, stream>>>(feats[L], wbf, conv_b, y); break;
    }
    k_project<<<(M + 255) / 256, 256, 0, stream>>>(y, score_w, score_b,
                                                   delta_w, delta_b, sc, dl, HW);
    k_topk_decode<<<2, 1024, 0, stream>>>(sc, dl, cat_boxes, cat_scores,
                                          n, K, offc, W, STR[L], SIZ[L]);
    offc += K;
  }

  k_sort<<<2, 1024, 0, stream>>>(cat_scores, cat_boxes, sorted_boxes);
  dim3 gm(NWORDS, NWORDS, 2);
  k_nms_mask<<<gm, 32, 0, stream>>>(sorted_boxes, mask);
  k_nms_scan<<<2, 256, 0, stream>>>(sorted_boxes, mask, out);
}
